// MoEFusion_52647709114885
// MI455X (gfx1250) — matrix operand order repaired and checked by reading, not compiled, not run on hardware
//
#include <hip/hip_runtime.h>
#include <hip/hip_bf16.h>
#include <math.h>

// ---------------------------------------------------------------------------
// MoE fusion for MI455X (gfx1250). Compute-bound (~260 GFLOP after pruning
// softmax-over-1-key identities); GEMMs use v_wmma_f32_16x16x32_bf16 with
// fp32 accumulation, fed by double-buffered GLOBAL_LOAD_ASYNC_TO_LDS_B128
// (ASYNCcnt) so global latency hides behind the WMMA pipeline.
// ---------------------------------------------------------------------------

typedef __bf16 bf16_t;
typedef __attribute__((ext_vector_type(16))) __bf16 v16bf;
typedef __attribute__((ext_vector_type(8)))  __bf16 v8bf;
typedef __attribute__((ext_vector_type(8)))  float  v8f;

#define BN_BATCH 16384
#define DMODEL   512
#define NHEAD    8
#define HDIM     64

#define BM 128
#define BN 128
#define BK 32
#define LDSA 40   // padded bf16 row stride (32 + 8) -> conflict-free 16B frags
#define LDSB 40

__device__ __forceinline__ float gelu_exact(float x) {
    return 0.5f * x * (1.0f + erff(x * 0.70710678118654752f));
}

// Async global->LDS copy of 16 bytes per lane (CDNA5 GLOBAL_LOAD_ASYNC_TO_LDS,
// tracked by ASYNCcnt; no VGPR round-trip, no loadcnt stall).
__device__ __forceinline__ void async_cp16(void* lds, const void* gsrc) {
    unsigned ldsaddr = (unsigned)(unsigned long long)(uintptr_t)lds;
    asm volatile("global_load_async_to_lds_b128 %0, %1, off"
                 :: "v"(ldsaddr), "v"(gsrc) : "memory");
}
__device__ __forceinline__ void wait_async0() {
    asm volatile("s_wait_asynccnt 0x0" ::: "memory");
}

// ---------------------------------------------------------------------------
// GEMM: C[M,N] = A[M,K](bf16 row-major) @ W[N,K](bf16 row-major)^T + bias,
// optional exact GELU; writes fp32 and/or bf16 with row stride ldc.
// 128x128x32 tiles, 8 waves, each wave: 2 M-frags x 4 N-frags = 8 WMMA/K-step.
// Double-buffered LDS filled by async copies issued one K-step ahead.
// ---------------------------------------------------------------------------
__global__ __launch_bounds__(256) void gemm_bf16_wmma(
    const bf16_t* __restrict__ A, const bf16_t* __restrict__ W,
    const float* __restrict__ bias, float* __restrict__ Cf,
    bf16_t* __restrict__ Cb, int M, int N, int K, int ldc, int act)
{
    __shared__ bf16_t lA[2][BM * LDSA];
    __shared__ bf16_t lB[2][BN * LDSB];

    const int tid  = threadIdx.x;
    const int wave = tid >> 5;
    const int lane = tid & 31;
    const int l15  = lane & 15;
    const int half = (lane < 16) ? 0 : 1;
    const int mw   = wave & 3;          // M group: 32 rows each
    const int nw   = wave >> 2;         // N group: 64 cols each
    const int bm = blockIdx.x * BM;
    const int bn = blockIdx.y * BN;

    // Staging coordinates: 512 x 16B chunks per 128x32 tile, 2 per thread.
    const int sr0 = tid >> 1;                  // rows 0..127
    const int sc0 = (tid & 1) * 16;            // cols 0 / 16
    // (two chunks per thread: columns sc0 and sc0+8)

    v8f acc[2][4];
    #pragma unroll
    for (int mt = 0; mt < 2; ++mt)
        #pragma unroll
        for (int nt = 0; nt < 4; ++nt)
            #pragma unroll
            for (int i = 0; i < 8; ++i) acc[mt][nt][i] = 0.0f;

    auto stage = [&](int buf, int k0) {
        // A tile 128x32
        #pragma unroll
        for (int i = 0; i < 2; ++i) {
            int c = sc0 + i * 8;
            async_cp16(&lA[buf][sr0 * LDSA + c],
                       A + (size_t)(bm + sr0) * K + k0 + c);
        }
        // W tile 128x32 (clamp rows >= N to row N-1: valid reads, the
        // affected output columns are never stored)
        int wr = bn + sr0; if (wr >= N) wr = N - 1;
        #pragma unroll
        for (int i = 0; i < 2; ++i) {
            int c = sc0 + i * 8;
            async_cp16(&lB[buf][sr0 * LDSB + c],
                       W + (size_t)wr * K + k0 + c);
        }
    };

    const int nk = K / BK;
    stage(0, 0);
    wait_async0();
    __syncthreads();

    for (int kk = 0; kk < nk; ++kk) {
        const int cur = kk & 1;
        if (kk + 1 < nk) stage(1 - cur, (kk + 1) * BK);

        const bf16_t* la = lA[cur];
        const bf16_t* lb = lB[cur];

        // A fragments per ISA layout: lane<16 -> M=lane, K={0..7,16..23};
        //                             lane>=16 -> M=lane-16, K={8..15,24..31}.
        v16bf af[2];
        const int kb = half * 8;
        #pragma unroll
        for (int mt = 0; mt < 2; ++mt) {
            int m = (mw * 2 + mt) * 16 + l15;
            v8bf lo = *(const v8bf*)&la[m * LDSA + kb];
            v8bf hi = *(const v8bf*)&la[m * LDSA + kb + 16];
            #pragma unroll
            for (int i = 0; i < 8; ++i) { af[mt][i] = lo[i]; af[mt][8 + i] = hi[i]; }
        }
        const int kh = half * 8;
        #pragma unroll
        for (int nt = 0; nt < 4; ++nt) {
            // B fragment: lane<16 -> col=lane, K={0..7,16..23}; lane>=16 -> K={8..15,24..31}.
            int n = (nw * 4 + nt) * 16 + l15;
            v8bf blo = *(const v8bf*)&lb[n * LDSB + kh];
            v8bf bhi = *(const v8bf*)&lb[n * LDSB + kh + 16];
            v16bf bfrag;
            #pragma unroll
            for (int i = 0; i < 8; ++i) { bfrag[i] = blo[i]; bfrag[8 + i] = bhi[i]; }
            #pragma unroll
            for (int mt = 0; mt < 2; ++mt)
                acc[mt][nt] = __builtin_amdgcn_wmma_f32_16x16x32_bf16(
                    false, af[mt], false, bfrag, (short)0, acc[mt][nt], false, false);
        }

        wait_async0();     // our async fills for the next buffer are done
        __syncthreads();   // everyone finished reading cur / writing next
    }

    // Epilogue. C layout: VGPR i -> (M = i + 8*half, N = lane&15).
    #pragma unroll
    for (int nt = 0; nt < 4; ++nt) {
        int col = bn + (nw * 4 + nt) * 16 + l15;
        if (col >= N) continue;
        float bv = bias ? bias[col] : 0.0f;
        #pragma unroll
        for (int mt = 0; mt < 2; ++mt) {
            #pragma unroll
            for (int i = 0; i < 8; ++i) {
                int row = bm + (mw * 2 + mt) * 16 + i + 8 * half;
                float v = acc[mt][nt][i] + bv;
                if (act == 1) v = gelu_exact(v);
                if (Cf) Cf[(size_t)row * ldc + col] = v;
                if (Cb) Cb[(size_t)row * ldc + col] = (bf16_t)v;
            }
        }
    }
}

// ---------------------------------------------------------------------------
// out = alpha*(a [+ b]); optional fp32 and/or bf16 outputs. Also the
// fp32->bf16 converter (b=null, alpha=1).
// ---------------------------------------------------------------------------
__global__ __launch_bounds__(256) void addcvt_kernel(
    const float* __restrict__ a, const float* __restrict__ b, float alpha,
    float* __restrict__ of, bf16_t* __restrict__ ob, int n)
{
    int i = blockIdx.x * 256 + threadIdx.x;
    if (i >= n) return;
    float v = a[i];
    if (b) v += b[i];
    v *= alpha;
    if (of) of[i] = v;
    if (ob) ob[i] = (bf16_t)v;
}

// ---------------------------------------------------------------------------
// y = LayerNorm((R? R+Y : Y)) * g + b.   Wave-per-row, shuffle reductions.
// ---------------------------------------------------------------------------
__global__ __launch_bounds__(256) void resln_kernel(
    const float* __restrict__ R, const float* __restrict__ Y,
    const float* __restrict__ g, const float* __restrict__ b,
    float* __restrict__ outF, bf16_t* __restrict__ outB, int rows)
{
    const int wave = threadIdx.x >> 5;
    const int lane = threadIdx.x & 31;
    const int row  = blockIdx.x * 8 + wave;
    if (row >= rows) return;
    const float* y = Y + (size_t)row * DMODEL;
    const float* r = R ? R + (size_t)row * DMODEL : nullptr;

    float x[DMODEL / 32];
    float s = 0.0f;
    #pragma unroll
    for (int i = 0; i < DMODEL / 32; ++i) {
        float v = y[lane + i * 32];
        if (r) v += r[lane + i * 32];
        x[i] = v; s += v;
    }
    #pragma unroll
    for (int o = 16; o >= 1; o >>= 1) s += __shfl_xor(s, o, 32);
    float mean = s * (1.0f / DMODEL);
    float vs = 0.0f;
    #pragma unroll
    for (int i = 0; i < DMODEL / 32; ++i) { float d = x[i] - mean; vs += d * d; }
    #pragma unroll
    for (int o = 16; o >= 1; o >>= 1) vs += __shfl_xor(vs, o, 32);
    float inv = rsqrtf(vs * (1.0f / DMODEL) + 1e-5f);
    #pragma unroll
    for (int i = 0; i < DMODEL / 32; ++i) {
        int c = lane + i * 32;
        float o = (x[i] - mean) * inv * g[c] + b[c];
        if (outF) outF[(size_t)row * DMODEL + c] = o;
        if (outB) outB[(size_t)row * DMODEL + c] = (bf16_t)o;
    }
}

// ---------------------------------------------------------------------------
// Cat-expert attention: seq len 2 (tokens img/txt), per-(row,head) 2-key
// softmax. Outputs bf16 (feeds Wo GEMM). ~0.1 GFLOP total - negligible.
// ---------------------------------------------------------------------------
__global__ __launch_bounds__(256) void attn2_kernel(
    const float* __restrict__ Qi, const float* __restrict__ Qt,
    const float* __restrict__ Ki, const float* __restrict__ Kt,
    const float* __restrict__ Vi, const float* __restrict__ Vt,
    bf16_t* __restrict__ Oi, bf16_t* __restrict__ Ot, int rows)
{
    int idx = blockIdx.x * 256 + threadIdx.x;
    if (idx >= rows * NHEAD) return;
    int row = idx / NHEAD, h = idx % NHEAD;
    size_t base = (size_t)row * DMODEL + h * HDIM;
    const float *qi = Qi + base, *qt = Qt + base, *ki = Ki + base,
                *kt = Kt + base, *vi = Vi + base, *vt = Vt + base;
    float sii = 0, sit = 0, sti = 0, stt = 0;
    #pragma unroll 8
    for (int d = 0; d < HDIM; ++d) {
        sii += qi[d] * ki[d]; sit += qi[d] * kt[d];
        sti += qt[d] * ki[d]; stt += qt[d] * kt[d];
    }
    const float sc = 0.125f;  // 1/sqrt(64)
    sii *= sc; sit *= sc; sti *= sc; stt *= sc;
    float m0 = fmaxf(sii, sit), e00 = __expf(sii - m0), e01 = __expf(sit - m0);
    float a0 = e00 / (e00 + e01), a1 = 1.0f - a0;
    float m1 = fmaxf(sti, stt), e10 = __expf(sti - m1), e11 = __expf(stt - m1);
    float b0 = e10 / (e10 + e11), b1 = 1.0f - b0;
    #pragma unroll 8
    for (int d = 0; d < HDIM; ++d) {
        Oi[base + d] = (bf16_t)(a0 * vi[d] + a1 * vt[d]);
        Ot[base + d] = (bf16_t)(b0 * vi[d] + b1 * vt[d]);
    }
}

// gates = softmax(logits) (TAU=1); also written to d_out tail.
__global__ __launch_bounds__(256) void gates_kernel(
    const float* __restrict__ logits, float* __restrict__ gates,
    float* __restrict__ outTail, int rows)
{
    int row = blockIdx.x * 256 + threadIdx.x;
    if (row >= rows) return;
    float l0 = logits[row * 3], l1 = logits[row * 3 + 1], l2 = logits[row * 3 + 2];
    float m = fmaxf(l0, fmaxf(l1, l2));
    float e0 = __expf(l0 - m), e1 = __expf(l1 - m), e2 = __expf(l2 - m);
    float inv = 1.0f / (e0 + e1 + e2);
    float g0 = e0 * inv, g1 = e1 * inv, g2 = e2 * inv;
    gates[row * 3] = g0; gates[row * 3 + 1] = g1; gates[row * 3 + 2] = g2;
    outTail[row * 3] = g0; outTail[row * 3 + 1] = g1; outTail[row * 3 + 2] = g2;
}

__global__ __launch_bounds__(256) void fuse_kernel(
    const float* __restrict__ gates, const float* __restrict__ catO,
    const float* __restrict__ addO, const float* __restrict__ crsO,
    float* __restrict__ out, int n)
{
    int i = blockIdx.x * 256 + threadIdx.x;
    if (i >= n) return;
    int row = i / DMODEL;
    out[i] = gates[row * 3] * catO[i] + gates[row * 3 + 1] * addO[i]
           + gates[row * 3 + 2] * crsO[i];
}

// ---------------------------------------------------------------------------
// Host orchestration.
// Input flattening assumed in setup_inputs() dict traversal order:
//  0 img, 1 txt, 2..17 cat{Wq,Wk,Wv,Wo,bq,bk,bv,bo,W1,b1,W2,b2,ln1g,ln1b,ln2g,ln2b},
//  18..33 add (same layout), 34..53 cross{i2t(8), t2i(8), projW, projb, lng, lnb},
//  54..63 router{Wi,bi,gi,bgi,Wt,bt,gt,bgt,Wo,bo}.
// ---------------------------------------------------------------------------
extern "C" void kernel_launch(void* const* d_in, const int* in_sizes, int n_in,
                              void* d_out, int out_size, void* d_ws, size_t ws_size,
                              hipStream_t stream)
{
    const int Bq = BN_BATCH, Dm = DMODEL, D2 = 2 * DMODEL;
    const size_t BD = (size_t)Bq * Dm;
    const float* img = (const float*)d_in[0];
    const float* txt = (const float*)d_in[1];
    auto F32 = [&](int i) { return (const float*)d_in[i]; };

    uint8_t* wsp = (uint8_t*)d_ws;
    size_t off = 0;
    auto alloc = [&](size_t bytes) -> void* {
        void* p = wsp + off;
        off = (off + bytes + 255) & ~(size_t)255;
        return p;
    };

    auto cvt = [&](const float* src, bf16_t* dst, size_t n) {
        addcvt_kernel<<<(unsigned)((n + 255) / 256), 256, 0, stream>>>(
            src, nullptr, 1.0f, nullptr, dst, (int)n);
    };
    auto cvtW = [&](int idx, size_t n) -> bf16_t* {
        bf16_t* p = (bf16_t*)alloc(n * sizeof(bf16_t));
        cvt(F32(idx), p, n);
        return p;
    };
    auto gemm = [&](const bf16_t* A, const bf16_t* Wb, const float* bias,
                    float* Cf, bf16_t* Cb, int N, int K, int ldc, int act) {
        dim3 grid(Bq / BM, (N + BN - 1) / BN);
        gemm_bf16_wmma<<<grid, 256, 0, stream>>>(A, Wb, bias, Cf, Cb, Bq, N, K, ldc, act);
    };
    auto resln = [&](const float* R, const float* Y, int gi, int bi,
                     float* oF, bf16_t* oB) {
        resln_kernel<<<Bq / 8, 256, 0, stream>>>(R, Y, F32(gi), F32(bi), oF, oB, Bq);
    };

    // --- scratch ---
    float* F[8];
    for (int i = 0; i < 8; ++i) F[i] = (float*)alloc(BD * 4);
    bf16_t* ib   = (bf16_t*)alloc(BD * 2);
    bf16_t* tb   = (bf16_t*)alloc(BD * 2);
    bf16_t* ab   = (bf16_t*)alloc(BD * 2);
    bf16_t* vb   = (bf16_t*)alloc(BD * 2);
    bf16_t* x1b  = (bf16_t*)alloc(BD * 2);
    bf16_t* oib  = (bf16_t*)alloc(BD * 2);
    bf16_t* otb  = (bf16_t*)alloc(BD * 2);
    bf16_t* hsb  = (bf16_t*)alloc(BD * 2);
    bf16_t* hb   = (bf16_t*)alloc(BD * 2 * 2);   // B x 2D
    bf16_t* ccat = (bf16_t*)alloc(BD * 2 * 2);   // B x 2D concat(oi,ot)
    float*  logits = (float*)alloc((size_t)Bq * 3 * 4);
    float*  gates  = (float*)alloc((size_t)Bq * 3 * 4);

    // --- weights -> bf16 (layout N x K row-major, exactly W as stored) ---
    const size_t DD = (size_t)Dm * Dm, DD2 = (size_t)Dm * D2;
    bf16_t* wCq = cvtW(2, DD);  bf16_t* wCk = cvtW(3, DD);
    bf16_t* wCv = cvtW(4, DD);  bf16_t* wCo = cvtW(5, DD);
    bf16_t* wC1 = cvtW(10, DD2); bf16_t* wC2 = cvtW(12, DD2);
    bf16_t* wAv = cvtW(20, DD); bf16_t* wAo = cvtW(21, DD);
    bf16_t* wA1 = cvtW(26, DD2); bf16_t* wA2 = cvtW(28, DD2);
    bf16_t* wXv = cvtW(36, DD); bf16_t* wXo = cvtW(37, DD);   // i2t Wv, Wo
    bf16_t* wYv = cvtW(44, DD); bf16_t* wYo = cvtW(45, DD);   // t2i Wv, Wo
    bf16_t* wPj = cvtW(50, DD2);
    bf16_t* wRi = cvtW(54, DD); bf16_t* wRt = cvtW(58, DD);
    bf16_t* wRo = cvtW(62, (size_t)3 * Dm);

    cvt(img, ib, BD);
    cvt(txt, tb, BD);

    // ---- Router: hi=LN(gelu(img@Wi+bi)); ht likewise; logits=(hi+ht)@Wo+bo
    gemm(ib, wRi, F32(55), F[0], nullptr, Dm, Dm, Dm, 1);
    resln(nullptr, F[0], 56, 57, F[1], nullptr);
    gemm(tb, wRt, F32(59), F[0], nullptr, Dm, Dm, Dm, 1);
    resln(nullptr, F[0], 60, 61, F[2], nullptr);
    addcvt_kernel<<<(unsigned)((BD + 255) / 256), 256, 0, stream>>>(
        F[1], F[2], 1.0f, nullptr, hsb, (int)BD);
    gemm(hsb, wRo, F32(63), logits, nullptr, 3, Dm, 3, 0);
    gates_kernel<<<Bq / 256, 256, 0, stream>>>(logits, gates,
                                               (float*)d_out + BD, Bq);

    // ---- Cross expert (softmax over 1 key == identity -> skip Q/K):
    // oi = (txt@Wv_i2t+bv)@Wo_i2t+bo ; ot = (img@Wv_t2i+bv)@Wo_t2i+bo
    gemm(tb, wXv, F32(40), nullptr, vb, Dm, Dm, Dm, 0);
    gemm(vb, wXo, F32(41), nullptr, ccat, Dm, Dm, D2, 0);          // cols [0,512)
    gemm(ib, wYv, F32(48), nullptr, vb, Dm, Dm, Dm, 0);
    gemm(vb, wYo, F32(49), nullptr, ccat + Dm, Dm, Dm, D2, 0);     // cols [512,1024)
    gemm(ccat, wPj, F32(51), F[0], nullptr, Dm, D2, Dm, 0);
    resln(nullptr, F[0], 52, 53, F[3], nullptr);                   // F3 = cross_out

    // ---- Add expert (same 1-key identity): x = img+txt
    addcvt_kernel<<<(unsigned)((BD + 255) / 256), 256, 0, stream>>>(
        img, txt, 1.0f, F[0], ab, (int)BD);
    gemm(ab, wAv, F32(24), nullptr, vb, Dm, Dm, Dm, 0);
    gemm(vb, wAo, F32(25), F[1], nullptr, Dm, Dm, Dm, 0);
    resln(F[0], F[1], 30, 31, F[2], x1b);                          // x1
    gemm(x1b, wA1, F32(27), nullptr, hb, D2, Dm, D2, 1);           // gelu
    gemm(hb, wA2, F32(29), F[1], nullptr, Dm, D2, Dm, 0);
    resln(F[2], F[1], 32, 33, F[4], nullptr);                      // F4 = add_out

    // ---- Cat expert (seq len 2: real attention)
    gemm(ib, wCq, F32(6), F[0], nullptr, Dm, Dm, Dm, 0);  // Qi
    gemm(tb, wCq, F32(6), F[1], nullptr, Dm, Dm, Dm, 0);  // Qt
    gemm(ib, wCk, F32(7), F[2], nullptr, Dm, Dm, Dm, 0);  // Ki
    gemm(tb, wCk, F32(7), F[5], nullptr, Dm, Dm, Dm, 0);  // Kt
    gemm(ib, wCv, F32(8), F[6], nullptr, Dm, Dm, Dm, 0);  // Vi
    gemm(tb, wCv, F32(8), F[7], nullptr, Dm, Dm, Dm, 0);  // Vt
    attn2_kernel<<<(Bq * NHEAD) / 256, 256, 0, stream>>>(
        F[0], F[1], F[2], F[5], F[6], F[7], oib, otb, Bq);
    gemm(oib, wCo, F32(9), F[0], nullptr, Dm, Dm, Dm, 0); // attn out (img)
    gemm(otb, wCo, F32(9), F[1], nullptr, Dm, Dm, Dm, 0); // attn out (txt)
    // token img
    resln(img, F[0], 14, 15, F[2], x1b);
    gemm(x1b, wC1, F32(11), nullptr, hb, D2, Dm, D2, 1);
    gemm(hb, wC2, F32(13), F[0], nullptr, Dm, D2, Dm, 0);
    resln(F[2], F[0], 16, 17, F[5], nullptr);              // x2 img
    // token txt
    resln(txt, F[1], 14, 15, F[2], x1b);
    gemm(x1b, wC1, F32(11), nullptr, hb, D2, Dm, D2, 1);
    gemm(hb, wC2, F32(13), F[0], nullptr, Dm, D2, Dm, 0);
    resln(F[2], F[0], 16, 17, F[6], nullptr);              // x2 txt
    addcvt_kernel<<<(unsigned)((BD + 255) / 256), 256, 0, stream>>>(
        F[5], F[6], 0.5f, F[7], nullptr, (int)BD);         // F7 = cat_out (mean)

    // ---- Gated fuse -> d_out[0, B*D)
    fuse_kernel<<<(unsigned)((BD + 255) / 256), 256, 0, stream>>>(
        gates, F[7], F[4], F[3], (float*)d_out, (int)BD);
}
